// DenseGraphConvNodeToEdge_41583873360239
// MI455X (gfx1250) — matrix operand order repaired and checked by reading, not compiled, not run on hardware
//
#include <hip/hip_runtime.h>

typedef __attribute__((ext_vector_type(16))) _Float16 v16h;
typedef __attribute__((ext_vector_type(8)))  float    v8f;
typedef __attribute__((ext_vector_type(4)))  float    f4;

constexpr int kB = 4, kN = 1024, kC = 64;

// ---------------------------------------------------------------------------
// Kernel A: xs[b,c] = sum_n x[b,n,c];  T[b,o] = bias[o] + sum_c xs[b,c]*W2[o,c]
// ---------------------------------------------------------------------------
__global__ void colsum_w2_bias(const float* __restrict__ x,
                               const float* __restrict__ W2,
                               const float* __restrict__ bias,
                               float* __restrict__ T) {
  __shared__ float part[4][64];
  __shared__ float xs[64];
  const int b = blockIdx.x;          // 0..3
  const int t = threadIdx.x;         // 0..255
  const int c = t & 63;
  const int p = t >> 6;              // 0..3, n-chunk
  const float* xb = x + (size_t)b * kN * kC;
  float s = 0.f;
  #pragma unroll 4
  for (int n = p * 256; n < (p + 1) * 256; ++n) s += xb[(size_t)n * kC + c];
  part[p][c] = s;
  __syncthreads();
  if (t < 64) xs[t] = part[0][t] + part[1][t] + part[2][t] + part[3][t];
  __syncthreads();
  if (t < 64) {                      // t == o
    float acc = bias[t];             // bias shape (64,1) -> flat [o]
    const float* w = W2 + (size_t)t * kC;
    #pragma unroll
    for (int cc = 0; cc < kC; ++cc) acc = fmaf(xs[cc], w[cc], acc);
    T[b * kC + t] = acc;
  }
}

// ---------------------------------------------------------------------------
// Kernel B: WMMA. Per wave: one 16-row tile of flat X (4096x64).
//   Y0[n,o] = sum_c X[n,c]*W0[o,c]
//   Y1[n,o] = sum_c X[n,c]*W1[o,c] + T[b(n),o]
// A-layout (f16 16x32): lane<16 -> M=lane, element h -> K = h;
//                       lanes 16-31 add +16 to K.
// B-layout (f16 32x16): lane&15 = N(=o); half h -> K = h (+16 for lanes 16-31).
// ---------------------------------------------------------------------------
__global__ void gemm_wmma(const float* __restrict__ x,
                          const float* __restrict__ W0,
                          const float* __restrict__ W1,
                          const float* __restrict__ T,
                          float* __restrict__ Y0,
                          float* __restrict__ Y1) {
  const int lane = threadIdx.x & 31;
  const int wid  = threadIdx.x >> 5;
  const int tile = blockIdx.x * 8 + wid;   // 0..255 : rows tile*16 .. tile*16+15
  const int m    = lane & 15;
  const bool hi  = lane >= 16;

  v8f acc0[4] = {};
  v8f acc1[4] = {};

  #pragma unroll
  for (int kt = 0; kt < 2; ++kt) {
    const int kbase = kt * 32;
    // ---- A tile: row m of this 16-row slab, K = kbase..kbase+31 ----
    v16h a;
    const float* xrow = x + (size_t)(tile * 16 + m) * kC + kbase;
    const int khalf = hi ? 16 : 0;
    #pragma unroll
    for (int h = 0; h < 16; ++h) {
      const int v  = h >> 1;
      const int pb = h & 1;
      const int k  = khalf + h;
      a[h] = (_Float16)xrow[k];
    }
    const int koff = hi ? 16 : 0;
    #pragma unroll
    for (int ot = 0; ot < 4; ++ot) {
      const int o = ot * 16 + m;                       // output column
      const float* w0row = W0 + (size_t)o * kC + kbase + koff;
      const float* w1row = W1 + (size_t)o * kC + kbase + koff;
      v16h b0, b1;
      #pragma unroll
      for (int h = 0; h < 16; ++h) {
        b0[h] = (_Float16)w0row[h];
        b1[h] = (_Float16)w1row[h];
      }
      acc0[ot] = __builtin_amdgcn_wmma_f32_16x16x32_f16(
          false, a, false, b0, (short)0, acc0[ot], false, false);
      acc1[ot] = __builtin_amdgcn_wmma_f32_16x16x32_f16(
          false, a, false, b1, (short)0, acc1[ot], false, false);
    }
  }

  // C/D layout: lane&15 = N(o), VGPR r = M row (+8 for lanes 16-31)
  const int b    = (tile * 16) >> 10;          // batch (1024 % 16 == 0 -> uniform per tile)
  const int mrow = tile * 16 + (hi ? 8 : 0);
  #pragma unroll
  for (int ot = 0; ot < 4; ++ot) {
    const int o = ot * 16 + m;
    const float tb = T[b * kC + o];
    #pragma unroll
    for (int r = 0; r < 8; ++r) {
      Y0[(size_t)(mrow + r) * kC + o] = acc0[ot][r];
      Y1[(size_t)(mrow + r) * kC + o] = acc1[ot][r] + tb;
    }
  }
}

// ---------------------------------------------------------------------------
// Kernel C: out[b,i,j,o] = Y0[b,j,o] + Y1'[b,i,o]   (1 GiB streaming store)
// Block = fixed (b,i), 64 consecutive j. Wave writes 512B contiguous per iter.
// ---------------------------------------------------------------------------
__global__ void write_out(const float* __restrict__ Y0,
                          const float* __restrict__ Y1,
                          float* __restrict__ out) {
  const int blk  = blockIdx.x;        // 65536
  const int jblk = blk & 15;          // 16 j-blocks of 64
  const int bi   = blk >> 4;          // flat (b,i) 0..4095
  const int b    = bi >> 10;
  const int t    = threadIdx.x;       // 256
  const int o4   = (t & 15) << 2;     // float4 offset in o
  const int j0   = jblk * 64 + (t >> 4);

  const f4 r = *(const f4*)(Y1 + (size_t)bi * kC + o4);
  float* obase       = out + (size_t)bi * kN * kC;
  const float* cbase = Y0 + (size_t)b * kN * kC;

  #pragma unroll
  for (int it = 0; it < 4; ++it) {
    const int j = j0 + it * 16;
    const f4 c = *(const f4*)(cbase + (size_t)j * kC + o4);
    f4 v = r + c;
    __builtin_nontemporal_store(v, (f4*)(obase + (size_t)j * kC + o4));
  }
}

// ---------------------------------------------------------------------------
extern "C" void kernel_launch(void* const* d_in, const int* in_sizes, int n_in,
                              void* d_out, int out_size, void* d_ws, size_t ws_size,
                              hipStream_t stream) {
  const float* x    = (const float*)d_in[0];
  // d_in[1] = adj (unused by forward)
  const float* W0   = (const float*)d_in[2];
  const float* W1   = (const float*)d_in[3];
  const float* W2   = (const float*)d_in[4];
  const float* bias = (const float*)d_in[5];
  float* out = (float*)d_out;

  float* Y0 = (float*)d_ws;                    // 4096*64 fp32 = 1 MB
  float* Y1 = Y0 + (size_t)kB * kN * kC;       // 1 MB
  float* T  = Y1 + (size_t)kB * kN * kC;       // 4*64 fp32

  colsum_w2_bias<<<kB, 256, 0, stream>>>(x, W2, bias, T);
  gemm_wmma<<<32, 256, 0, stream>>>(x, W0, W1, T, Y0, Y1);   // 256 tiles / 8 waves
  write_out<<<kB * kN * (kN / 64), 256, 0, stream>>>(Y0, Y1, out);
}
